// FPN_24395414241367
// MI455X (gfx1250) — matrix operand order repaired and checked by reading, not compiled, not run on hardware
//
#include <hip/hip_runtime.h>
#include <hip/hip_bf16.h>
#include <math.h>

// ---------------------------------------------------------------------------
// 3D RPN head, fused for MI455X (gfx1250, wave32, WMMA 16x16x32 bf16).
//   kernel 1: prep    - swizzle weights into WMMA A-fragment layout (bf16)
//   kernel 2: rpn     - conv3x3x3+ReLU -> two 1x1 GEMMs -> softmax/box decode
//                       (feat kept entirely in LDS as bf16 B-fragments)
//   kernel 3: nms     - 300-step greedy NMS, single persistent block,
//                       suppress pass fused with next argmax pass
// ---------------------------------------------------------------------------

typedef __attribute__((ext_vector_type(16))) __bf16 v16bf;
typedef __attribute__((ext_vector_type(8)))  float  v8f;

#define HF 28
#define WF 28
#define LF 112
#define NANCH 9
#define PPOS (HF * WF * LF)          // 87808
#define NTOT (NANCH * PPOS)          // 790272
#define MAXOUT 300

// A-fragment K index for 16-bit A 16x32 (contiguous, matches B fragments):
//   lanes 0-15 : e 0..15 -> K=e       (K 0..15)
//   lanes 16-31: e 0..15 -> K=e+16    (K 16..31)
__device__ __host__ inline int a_k_of(int lane, int e) {
    return e + (lane >= 16 ? 16 : 0);
}
// B-fragment (32x16, 16-bit): lane L -> N = L%16; e 0..15 -> K = e + (L>=16?16:0)

// ---------------------------------------------------------------------------
// Kernel 1: swizzle W_rpn [512,27] and W2 = [W_score(18,512); W_bbox(54,512)]
// into bf16 A-fragment layout.
//   gA1: [32 mTiles][32 lanes][16 e]            (K=27 padded to 32)
//   gA2: [5 mTiles][16 kChunks][32 lanes][16 e] (M=72 padded to 80)
// ---------------------------------------------------------------------------
__global__ __launch_bounds__(256) void prep_weights(
    const float* __restrict__ Wrpn, const float* __restrict__ Wsc,
    const float* __restrict__ Wbb, __bf16* __restrict__ gA1,
    __bf16* __restrict__ gA2) {
    int idx = blockIdx.x * 256 + threadIdx.x;
    const int N1 = 32 * 32 * 16;           // 16384
    const int N2 = 5 * 16 * 32 * 16;       // 40960
    if (idx < N1) {
        int e = idx & 15, lane = (idx >> 4) & 31, mT = idx >> 9;
        int c = mT * 16 + (lane & 15);
        int k = a_k_of(lane, e);
        float v = (k < 27) ? Wrpn[c * 27 + k] : 0.0f;
        gA1[idx] = (__bf16)v;
    } else if (idx < N1 + N2) {
        int i2 = idx - N1;
        int e = i2 & 15, lane = (i2 >> 4) & 31;
        int kc = (i2 >> 9) & 15, mT = i2 >> 13;
        int row = mT * 16 + (lane & 15);
        int kk = kc * 32 + a_k_of(lane, e);
        float v = 0.0f;
        if (row < 18)      v = Wsc[row * 512 + kk];
        else if (row < 72) v = Wbb[(row - 18) * 512 + kk];
        gA2[i2] = (__bf16)v;
    }
}

// ---------------------------------------------------------------------------
// Kernel 2: one block per (h,w). Handles all 112 l-positions (7 N-tiles).
// LDS plan (dynamic, ~164 KB of the 320 KB/WGP pool):
//   featB : 16 kChunks x 7 nTiles x 32 x 16 bf16  (stage-2 B fragments)
//   patchB: 7 nTiles x 32 x 16 bf16               (stage-1 im2col B frags)
//   out2  : 80 x 112 f32                          (stage-2 raw outputs)
//   xh    : 3 x 3 x 114 f32                       (x halo)
//   biases
// ---------------------------------------------------------------------------
__global__ __launch_bounds__(256) void rpn_fused(
    const float* __restrict__ x, const float* __restrict__ brpn,
    const float* __restrict__ bsc, const float* __restrict__ bbb,
    const __bf16* __restrict__ gA1, const __bf16* __restrict__ gA2,
    float* __restrict__ scores, float* __restrict__ alive,
    float* __restrict__ props) {
    extern __shared__ char smem[];
    __bf16* featB  = (__bf16*)smem;                  // 57344 bf16 = 114688 B
    __bf16* patchB = featB + 16 * 7 * 32 * 16;       // 3584 bf16  = 7168 B
    float*  out2   = (float*)(patchB + 3584);        // 8960 f32   = 35840 B
    float*  xh     = out2 + 80 * 112;                // 1026 f32
    float*  Lbrpn  = xh + 1026;                      // 512 f32
    float*  Lbsc   = Lbrpn + 512;                    // 18 f32
    float*  Lbbb   = Lbsc + 18;                      // 54 f32

    const int tid = threadIdx.x;
    const int bx  = blockIdx.x;
    const int h = bx / WF, w = bx % WF;

    // ---- phase 0: halo + biases into LDS ----
    for (int i = tid; i < 1026; i += 256) {
        int dh = i / 342, r0 = i % 342, dw = r0 / 114, j = r0 % 114;
        int gh = h + dh - 1, gw = w + dw - 1, gl = j - 1;
        float v = 0.0f;
        if (gh >= 0 && gh < HF && gw >= 0 && gw < WF && gl >= 0 && gl < LF)
            v = x[(gh * WF + gw) * LF + gl];
        xh[i] = v;
    }
    for (int i = tid; i < 512; i += 256) Lbrpn[i] = brpn[i];
    if (tid < 18) Lbsc[tid] = bsc[tid];
    if (tid < 54) Lbbb[tid] = bbb[tid];
    __syncthreads();

    // ---- phase 1: im2col patches pre-swizzled into B-fragment layout ----
    for (int i = tid; i < 7 * 32 * 16; i += 256) {
        int e = i & 15, lane = (i >> 4) & 31, nT = i >> 9;
        int t = e + (lane >= 16 ? 16 : 0);           // tap index = K
        float v = 0.0f;
        if (t < 27) {
            int dh = t / 9, dw = (t / 3) % 3, dl = t % 3;
            int l = nT * 16 + (lane & 15);
            v = xh[(dh * 3 + dw) * 114 + l + dl];
        }
        patchB[i] = (__bf16)v;
    }
    __syncthreads();

    const int wv = tid >> 5, lane = tid & 31;
    const int nCol = lane & 15;
    const int mBase = (lane >= 16) ? 8 : 0;

    // ---- phase 2: stage-1 GEMM (512 x 27pad32 x 112) + bias + ReLU,
    //      writeback as bf16 stage-2 B fragments ----
    for (int tile = wv; tile < 32 * 7; tile += 8) {
        int mT = tile / 7, nT = tile % 7;
        v16bf a = *(const v16bf*)(gA1 + (mT * 32 + lane) * 16);
        v16bf b = *(const v16bf*)(patchB + (nT * 32 + lane) * 16);
        v8f d = {};
        d = __builtin_amdgcn_wmma_f32_16x16x32_bf16(false, a, false, b,
                                                    (short)0, d, false, false);
#pragma unroll
        for (int r = 0; r < 8; ++r) {
            int c = mT * 16 + mBase + r;             // channel
            float v = d[r] + Lbrpn[c];
            v = v > 0.0f ? v : 0.0f;                 // ReLU
            int kc = c >> 5, within = c & 31;
            int half = within >> 4, e2 = within & 15;
            featB[(((kc * 7 + nT) * 32) + half * 16 + nCol) * 16 + e2] = (__bf16)v;
        }
    }
    __syncthreads();

    // ---- phase 3: stage-2 GEMM (80 x 512 x 112), accumulate over 16 K-chunks
    for (int tile = wv; tile < 5 * 7; tile += 8) {
        int mT = tile / 7, nT = tile % 7;
        v8f acc = {};
#pragma unroll 4
        for (int kc = 0; kc < 16; ++kc) {
            v16bf a = *(const v16bf*)(gA2 + ((mT * 16 + kc) * 32 + lane) * 16);
            v16bf b = *(const v16bf*)(featB + ((kc * 7 + nT) * 32 + lane) * 16);
            acc = __builtin_amdgcn_wmma_f32_16x16x32_bf16(false, a, false, b,
                                                          (short)0, acc, false, false);
        }
        int n = nT * 16 + nCol;
#pragma unroll
        for (int r = 0; r < 8; ++r) {
            int row = mT * 16 + mBase + r;
            if (row < 72) out2[row * 112 + n] = acc[r];
        }
    }
    __syncthreads();

    // ---- phase 4: softmax (2-way -> sigmoid), box decode, clip, scatter ----
    const float sc_tab[3] = {8.0f, 16.0f, 32.0f};
    const float sr_tab[3] = {0.70710678f, 1.0f, 1.41421356f};  // sqrt(ratio)
    for (int i = tid; i < LF * NANCH; i += 256) {
        int n = i % LF, k = i / LF;
        float bg = out2[k * 112 + n] + Lbsc[k];
        float fg = out2[(9 + k) * 112 + n] + Lbsc[9 + k];
        float score = 1.0f / (1.0f + expf(bg - fg));
        float d0 = out2[(18 + k * 6 + 0) * 112 + n] + Lbbb[k * 6 + 0];
        float d1 = out2[(18 + k * 6 + 1) * 112 + n] + Lbbb[k * 6 + 1];
        float d2 = out2[(18 + k * 6 + 2) * 112 + n] + Lbbb[k * 6 + 2];
        float d3 = out2[(18 + k * 6 + 3) * 112 + n] + Lbbb[k * 6 + 3];
        float d4 = out2[(18 + k * 6 + 4) * 112 + n] + Lbbb[k * 6 + 4];
        float d5 = out2[(18 + k * 6 + 5) * 112 + n] + Lbbb[k * 6 + 5];
        // anchors recomputed analytically (cx,cy,cz)=(w*16,h*16,l*4)
        float s = sc_tab[k / 3], sr = sr_tab[k % 3];
        float aw = s * 16.0f * sr, ah = s * 16.0f / sr, al = s * 4.0f;
        float cx = w * 16.0f, cy = h * 16.0f, cz = n * 4.0f;
        float px = d0 * aw + cx, py = d1 * ah + cy, pz = d2 * al + cz;
        float pw = expf(d3) * aw, ph = expf(d4) * ah, pl = expf(d5) * al;
        float x0 = fminf(fmaxf(px - pw * 0.5f, 0.0f), 447.0f);
        float y0 = fminf(fmaxf(py - ph * 0.5f, 0.0f), 447.0f);
        float z0 = fminf(fmaxf(pz - pl * 0.5f, 0.0f), 447.0f);
        float x1 = fminf(fmaxf(px + pw * 0.5f, 0.0f), 447.0f);
        float y1 = fminf(fmaxf(py + ph * 0.5f, 0.0f), 447.0f);
        float z1 = fminf(fmaxf(pz + pl * 0.5f, 0.0f), 447.0f);
        int g = k * PPOS + bx * LF + n;
        props[g * 6 + 0] = x0; props[g * 6 + 1] = y0; props[g * 6 + 2] = z0;
        props[g * 6 + 3] = x1; props[g * 6 + 4] = y1; props[g * 6 + 5] = z1;
        scores[g] = score;
        alive[g]  = score;
    }
}

// ---------------------------------------------------------------------------
// Kernel 3: greedy NMS, single persistent 1024-thread block, L2-resident.
// Suppress pass of step t is fused with argmax of step t+1.
// ---------------------------------------------------------------------------
__global__ __launch_bounds__(1024) void nms_kernel(
    const float* __restrict__ props, const float* __restrict__ scores,
    float* __restrict__ alive, float* __restrict__ out) {
    __shared__ float sval[1024];
    __shared__ int   sidx[1024];
    __shared__ int   chosen[MAXOUT];
    __shared__ float sbox[6];
    const int tid = threadIdx.x;
    const float NEG = -__builtin_inff();
    int cur = 0;

    for (int it = 0; it < MAXOUT; ++it) {
        if (it > 0 && tid < 6) sbox[tid] = props[cur * 6 + tid];
        __syncthreads();

        float bv = NEG;
        int   bi = 0;
        if (it == 0) {
            for (int g = tid; g < NTOT; g += 1024) {
                float v = alive[g];
                if (v > bv) { bv = v; bi = g; }
            }
        } else {
            float bx0 = sbox[0], by0 = sbox[1], bz0 = sbox[2];
            float bx1 = sbox[3], by1 = sbox[4], bz1 = sbox[5];
            float volb = (bx1 - bx0 + 1.0f) * (by1 - by0 + 1.0f) * (bz1 - bz0 + 1.0f);
            for (int g = tid; g < NTOT; g += 1024) {
                __builtin_prefetch(props + (size_t)(g + 2048) * 6, 0, 0);
                float a = alive[g];
                float px0 = props[g * 6 + 0], py0 = props[g * 6 + 1];
                float pz0 = props[g * 6 + 2], px1 = props[g * 6 + 3];
                float py1 = props[g * 6 + 4], pz1 = props[g * 6 + 5];
                float i0 = fmaxf(fminf(bx1, px1) - fmaxf(bx0, px0) + 1.0f, 0.0f);
                float i1 = fmaxf(fminf(by1, py1) - fmaxf(by0, py0) + 1.0f, 0.0f);
                float i2 = fmaxf(fminf(bz1, pz1) - fmaxf(bz0, pz0) + 1.0f, 0.0f);
                float inter = i0 * i1 * i2;
                float vg = (px1 - px0 + 1.0f) * (py1 - py0 + 1.0f) * (pz1 - pz0 + 1.0f);
                float iou = inter / (volb + vg - inter);
                if (iou > 0.7f || g == cur) { a = NEG; alive[g] = a; }
                if (a > bv) { bv = a; bi = g; }
            }
        }
        sval[tid] = bv; sidx[tid] = bi;
        __syncthreads();
        for (int s = 512; s > 0; s >>= 1) {
            if (tid < s) {
                float v2 = sval[tid + s]; int i2 = sidx[tid + s];
                if (v2 > sval[tid] || (v2 == sval[tid] && i2 < sidx[tid])) {
                    sval[tid] = v2; sidx[tid] = i2;
                }
            }
            __syncthreads();
        }
        cur = sidx[0];
        if (tid == 0) chosen[it] = cur;
        // next iteration's first __syncthreads guards chosen/sval reuse
    }
    __syncthreads();
    // rows: [batch=0 | box(6) | score]
    for (int r = tid; r < MAXOUT; r += 1024) {
        int g = chosen[r];
        out[r * 8 + 0] = 0.0f;
#pragma unroll
        for (int j = 0; j < 6; ++j) out[r * 8 + 1 + j] = props[g * 6 + j];
        out[r * 8 + 7] = scores[g];
    }
}

// ---------------------------------------------------------------------------
extern "C" void kernel_launch(void* const* d_in, const int* in_sizes, int n_in,
                              void* d_out, int out_size, void* d_ws, size_t ws_size,
                              hipStream_t stream) {
    const float* x    = (const float*)d_in[0];
    const float* Wrpn = (const float*)d_in[1];
    const float* brpn = (const float*)d_in[2];
    const float* Wsc  = (const float*)d_in[3];
    const float* bsc  = (const float*)d_in[4];
    const float* Wbb  = (const float*)d_in[5];
    const float* bbb  = (const float*)d_in[6];
    // d_in[7] = anchors: recomputed analytically on-device (saves 19 MB read)

    char* ws = (char*)d_ws;
    __bf16* gA1 = (__bf16*)ws;                              // 32768 B
    __bf16* gA2 = (__bf16*)(ws + 32768);                    // 81920 B
    float* scores = (float*)(ws + 114688);                  // NTOT f32
    float* alive  = scores + NTOT;                          // NTOT f32
    float* props  = alive + NTOT;                           // NTOT*6 f32

    prep_weights<<<224, 256, 0, stream>>>(Wrpn, Wsc, Wbb, gA1, gA2);

    // LDS: featB 114688 + patchB 7168 + out2 35840 + xh 4104 + biases 2336
    size_t smem = 16 * 7 * 32 * 16 * 2 + 3584 * 2 + 80 * 112 * 4 +
                  1026 * 4 + 512 * 4 + 18 * 4 + 54 * 4;     // = 164136 B
    rpn_fused<<<HF * WF, 256, smem, stream>>>(x, brpn, bsc, bbb, gA1, gA2,
                                              scores, alive, props);

    nms_kernel<<<1, 1024, 0, stream>>>(props, scores, alive, (float*)d_out);
}
